// Encoder1DBlock_10599979287063
// MI455X (gfx1250) — matrix operand order repaired and checked by reading, not compiled, not run on hardware
//
#include <hip/hip_runtime.h>
#include <hip/hip_bf16.h>

typedef __bf16 bf16_t;
typedef __attribute__((ext_vector_type(16))) __bf16 v16bf;
typedef __attribute__((ext_vector_type(8)))  __bf16 v8bf;
typedef __attribute__((ext_vector_type(4)))  __bf16 v4bf;
typedef __attribute__((ext_vector_type(8)))  float  v8f;

#define LN_EPS 1e-5f

// ---------------------------------------------------------------------------
// CDNA5 async global->LDS copy (ASYNCcnt-tracked, bypasses VGPRs).
// Generic LDS pointers carry the LDS byte offset in the low 32 bits
// (ISA 10.2), so truncation gives the DS address.
// ---------------------------------------------------------------------------
__device__ __forceinline__ void async_copy_b128(const bf16_t* g, bf16_t* l)
{
    unsigned           lds = (unsigned)(uintptr_t)l;
    unsigned long long ga  = (unsigned long long)(uintptr_t)g;
    asm volatile("global_load_async_to_lds_b128 %0, %1, off"
                 :: "v"(lds), "v"(ga) : "memory");
}
__device__ __forceinline__ void wait_async_0()
{
    asm volatile("s_wait_asynccnt 0x0" ::: "memory");
}

// ---------------------------------------------------------------------------
// f32 -> bf16 elementwise convert (weights / activations)
// ---------------------------------------------------------------------------
__global__ __launch_bounds__(256)
void k_cvt_bf16(const float* __restrict__ in, bf16_t* __restrict__ out, long n)
{
    long i = (long)blockIdx.x * 256 + threadIdx.x;
    if (i < n) out[i] = (bf16_t)in[i];
}

// ---------------------------------------------------------------------------
// LayerNorm over rows of length 768, output bf16. One workgroup per row.
// ---------------------------------------------------------------------------
__global__ __launch_bounds__(256)
void k_layernorm(const float* __restrict__ x, const float* __restrict__ g,
                 const float* __restrict__ b, bf16_t* __restrict__ y)
{
    const int D = 768;
    __shared__ float sa[256];
    __shared__ float sb[256];
    long row = blockIdx.x;
    const float* xr = x + row * D;
    int tid = threadIdx.x;
    float v[3];
    float s = 0.f, s2 = 0.f;
#pragma unroll
    for (int i = 0; i < 3; ++i) {
        v[i] = xr[tid + i * 256];
        s += v[i];
        s2 += v[i] * v[i];
    }
    sa[tid] = s; sb[tid] = s2;
    __syncthreads();
    for (int st = 128; st > 0; st >>= 1) {
        if (tid < st) { sa[tid] += sa[tid + st]; sb[tid] += sb[tid + st]; }
        __syncthreads();
    }
    float mean = sa[0] * (1.f / D);
    float var  = sb[0] * (1.f / D) - mean * mean;
    float inv  = rsqrtf(var + LN_EPS);
#pragma unroll
    for (int i = 0; i < 3; ++i) {
        int c = tid + i * 256;
        y[row * D + c] = (bf16_t)((v[i] - mean) * inv * g[c] + b[c]);
    }
}

// ---------------------------------------------------------------------------
// poly_norm over rows of length 1024 (scores). Newton solve for c such that
// sum 1/(-x-c)^2 == 1, then weights = (-x-c)^-2. One workgroup per row.
// ---------------------------------------------------------------------------
__global__ __launch_bounds__(256)
void k_poly_norm(const float* __restrict__ scores, bf16_t* __restrict__ w)
{
    const int S = 1024;
    __shared__ float sa[256];
    __shared__ float sb[256];
    long row = blockIdx.x;
    int tid = threadIdx.x;
    const float4* xr = (const float4*)(scores + row * (long)S);
    float4 xv = xr[tid];
    float x[4] = { xv.x, xv.y, xv.z, xv.w };

    float mx = fmaxf(fmaxf(x[0], x[1]), fmaxf(x[2], x[3]));
    sa[tid] = mx;
    __syncthreads();
    for (int st = 128; st > 0; st >>= 1) {
        if (tid < st) sa[tid] = fmaxf(sa[tid], sa[tid + st]);
        __syncthreads();
    }
    float c = -sa[0] - 1.0f;   // min(-x) - 1 ; guarantees base >= 1

    for (int it = 0; it < 6; ++it) {
        float s = 0.f, sd = 0.f;
#pragma unroll
        for (int i = 0; i < 4; ++i) {
            float bse = -x[i] - c;
            float r = 1.0f / bse;
            float r2 = r * r;
            s  += r2;        // base^-2
            sd += r2 * r;    // base^-3
        }
        __syncthreads();               // protect shared reuse
        sa[tid] = s; sb[tid] = sd;
        __syncthreads();
        for (int st = 128; st > 0; st >>= 1) {
            if (tid < st) { sa[tid] += sa[tid + st]; sb[tid] += sb[tid + st]; }
            __syncthreads();
        }
        c -= (sa[0] - 1.f) / (2.f * sb[0] + 1e-8f);
    }
    v4bf o;
#pragma unroll
    for (int i = 0; i < 4; ++i) {
        float bse = -x[i] - c;
        float r = 1.0f / bse;
        o[i] = (bf16_t)(r * r);
    }
    *(v4bf*)(w + row * (long)S + tid * 4) = o;
}

// ---------------------------------------------------------------------------
// Generic tiled bf16 WMMA GEMM:  C[M,N] = act( scale * A[M,K] @ B + bias + resid )
//   Tile: BM=128, BN=64, BK=64, double-buffered LDS, async global->LDS staging.
//   256 threads = 8 waves; each wave owns a 32x32 output patch
//   (2x2 tiles of V_WMMA_F32_16X16X32_BF16, 2 K-substeps per stage).
//   TRANSB:   B memory is [N,K] row-major (e.g. scores = q @ k^T)
//   batching: grid.z over (b,h) with independent strides for A/B/C.
// ---------------------------------------------------------------------------
template<bool TRANSB, bool GELU, bool HAS_RESID, bool OUT_BF16>
__global__ __launch_bounds__(256)
void k_gemm(const bf16_t* __restrict__ A, int lda, long aOffB, long aOffH,
            const bf16_t* __restrict__ Bm, int ldb, long bOffB, long bOffH,
            const float* __restrict__ bias,
            const float* __restrict__ resid,
            float* __restrict__ Cf, bf16_t* __restrict__ Cb,
            int ldc, long cOffB, long cOffH,
            int nH, int K, float scale)
{
    // +8 elem pad: 144B row stride (9 x 16B), keeps b128 alignment, avoids
    // systematic bank conflicts on fragment reads.
    __shared__ bf16_t As[2][128][72];
    __shared__ bf16_t Bs[2][64][72];

    const int tid    = threadIdx.x;
    const int lane   = tid & 31;
    const int wave   = tid >> 5;
    const int ln     = lane & 15;
    const int laneHi = lane >> 4;
    const int wm = (wave >> 1) * 32;        // 4 wave-rows
    const int wn = (wave & 1) * 32;         // 2 wave-cols
    const int tileM = blockIdx.y * 128;
    const int tileN = blockIdx.x * 64;
    const int bz = blockIdx.z;
    const int bb = bz / nH, hh = bz % nH;

    const bf16_t* Ap = A  + (long)bb * aOffB + (long)hh * aOffH;
    const bf16_t* Bp = Bm + (long)bb * bOffB + (long)hh * bOffH;
    const long cbase = (long)bb * cOffB + (long)hh * cOffH;

    v8f acc[2][2] = {};

    // ---- tile staging (async where the copy is contiguous) ----
    auto stageA = [&](int k0, int buf) {
#pragma unroll
        for (int i = 0; i < 4; ++i) {           // 128x64 tile: 4 x b128 / thread
            int c = tid + i * 256;
            int r = c >> 3, kc = (c & 7) * 8;
            async_copy_b128(Ap + (long)(tileM + r) * lda + (k0 + kc),
                            &As[buf][r][kc]);
        }
    };
    auto stageB = [&](int k0, int buf) {
        if constexpr (TRANSB) {                 // B is [N,K]: direct async copy
#pragma unroll
            for (int i = 0; i < 2; ++i) {       // 64x64 tile: 2 x b128 / thread
                int c = tid + i * 256;
                int r = c >> 3, kc = (c & 7) * 8;
                async_copy_b128(Bp + (long)(tileN + r) * ldb + (k0 + kc),
                                &Bs[buf][r][kc]);
            }
        } else {                                // B is [K,N]: transpose into LDS
#pragma unroll
            for (int i = 0; i < 2; ++i) {
                int c = tid + i * 256;
                int kr = c >> 3, nc = (c & 7) * 8;
                v8bf v = *(const v8bf*)(Bp + (long)(k0 + kr) * ldb + (tileN + nc));
#pragma unroll
                for (int j = 0; j < 8; ++j) Bs[buf][nc + j][kr] = v[j];
                if (k0 + 64 < K)                // hint next K-tile (global_prefetch_b8)
                    __builtin_prefetch(Bp + (long)(k0 + 64 + kr) * ldb + (tileN + nc), 0, 1);
            }
        }
    };

    // ---- prologue: stage first K-tile ----
    stageA(0, 0);
    stageB(0, 0);
    wait_async_0();
    __syncthreads();

    int cur = 0;
    for (int k0 = 0; k0 < K; k0 += 64) {
        const int nxt = cur ^ 1;
        if (k0 + 64 < K) {                      // overlap next stage with compute
            stageA(k0 + 64, nxt);
            stageB(k0 + 64, nxt);
        }

        // ---- compute on current buffers: 2 K-substeps of 32 ----
#pragma unroll
        for (int ks = 0; ks < 64; ks += 32) {
            // A 16x32 lane map: lanes0-15 row=ln K{0..7,16..23}; lanes16-31 K{8..15,24..31}
            v16bf af[2], bfr[2];
            const int aoff = laneHi * 8;
#pragma unroll
            for (int mi = 0; mi < 2; ++mi) {
                const bf16_t* p = &As[cur][wm + mi * 16 + ln][ks];
                v8bf a0 = *(const v8bf*)(p + aoff);
                v8bf a1 = *(const v8bf*)(p + aoff + 16);
                af[mi] = __builtin_shufflevector(a0, a1,
                          0,1,2,3,4,5,6,7,8,9,10,11,12,13,14,15);
            }
            // B 32x16 lane map: lane = N; lanes0-15 K{0..7,16..23}; lanes16-31 K{8..15,24..31}
            const int boff = laneHi * 8;
#pragma unroll
            for (int ni = 0; ni < 2; ++ni) {
                const bf16_t* p = &Bs[cur][wn + ni * 16 + ln][ks + boff];
                v8bf b0 = *(const v8bf*)(p);
                v8bf b1 = *(const v8bf*)(p + 16);
                bfr[ni] = __builtin_shufflevector(b0, b1,
                          0,1,2,3,4,5,6,7,8,9,10,11,12,13,14,15);
            }
#pragma unroll
            for (int mi = 0; mi < 2; ++mi)
#pragma unroll
                for (int ni = 0; ni < 2; ++ni)
                    acc[mi][ni] = __builtin_amdgcn_wmma_f32_16x16x32_bf16(
                        false, af[mi], false, bfr[ni], (short)0, acc[mi][ni],
                        false, false);
        }

        wait_async_0();        // next-stage async copies have landed in LDS
        __syncthreads();       // everyone done reading cur / writing nxt
        cur = nxt;
    }

    // ---- epilogue: C/D layout = VGPR r -> M = laneHi*8 + r, N = ln ----
#pragma unroll
    for (int mi = 0; mi < 2; ++mi)
#pragma unroll
        for (int ni = 0; ni < 2; ++ni) {
            int col = tileN + wn + ni * 16 + ln;
            float bvv = (bias != nullptr) ? bias[col] : 0.f;
#pragma unroll
            for (int r = 0; r < 8; ++r) {
                int rowg = tileM + wm + mi * 16 + laneHi * 8 + r;
                long idx = cbase + (long)rowg * ldc + col;
                float vv = acc[mi][ni][r] * scale + bvv;
                if constexpr (HAS_RESID) vv += resid[idx];
                if constexpr (GELU)      vv = 0.5f * vv * (1.f + erff(vv * 0.70710678118654752f));
                if constexpr (OUT_BF16)  Cb[idx] = (bf16_t)vv;
                else                     Cf[idx] = vv;
            }
        }
}

// ---------------------------------------------------------------------------
// host-side orchestration
// ---------------------------------------------------------------------------
extern "C" void kernel_launch(void* const* d_in, const int* in_sizes, int n_in,
                              void* d_out, int out_size, void* d_ws, size_t ws_size,
                              hipStream_t stream)
{
    (void)in_sizes; (void)n_in; (void)out_size; (void)ws_size;

    const float* x    = (const float*)d_in[0];
    const float* ln1g = (const float*)d_in[1];
    const float* ln1b = (const float*)d_in[2];
    const float* wq   = (const float*)d_in[3];
    const float* bq   = (const float*)d_in[4];
    const float* wk   = (const float*)d_in[5];
    const float* bk   = (const float*)d_in[6];
    const float* wv   = (const float*)d_in[7];
    const float* bv   = (const float*)d_in[8];
    const float* wo   = (const float*)d_in[9];
    const float* bo   = (const float*)d_in[10];
    const float* ln2g = (const float*)d_in[11];
    const float* ln2b = (const float*)d_in[12];
    const float* w1   = (const float*)d_in[13];
    const float* b1   = (const float*)d_in[14];
    const float* w2   = (const float*)d_in[15];
    const float* b2   = (const float*)d_in[16];

    const int B = 2, S = 1024, D = 768, H = 12, M4 = 3072;
    const long R = (long)B * S;                 // 2048 token rows

    // ---- carve workspace (~204 MB) ----
    char* wsb = (char*)d_ws;
    size_t off = 0;
    auto carve = [&](size_t bytes) -> void* {
        void* p = (void*)(wsb + off);
        off = (off + bytes + 255) & ~(size_t)255;
        return p;
    };
    bf16_t* wqb  = (bf16_t*)carve((size_t)D * D * 2);
    bf16_t* wkb  = (bf16_t*)carve((size_t)D * D * 2);
    bf16_t* wvb  = (bf16_t*)carve((size_t)D * D * 2);
    bf16_t* wob  = (bf16_t*)carve((size_t)D * D * 2);
    bf16_t* w1b  = (bf16_t*)carve((size_t)D * M4 * 2);
    bf16_t* w2b  = (bf16_t*)carve((size_t)M4 * D * 2);
    bf16_t* ylnb = (bf16_t*)carve((size_t)R * D * 2);
    bf16_t* qb   = (bf16_t*)carve((size_t)R * D * 2);
    bf16_t* kb   = (bf16_t*)carve((size_t)R * D * 2);
    bf16_t* vb   = (bf16_t*)carve((size_t)R * D * 2);
    bf16_t* attb = (bf16_t*)carve((size_t)R * D * 2);
    float*  x1   = (float*) carve((size_t)R * D * 4);
    bf16_t* y2b  = (bf16_t*)carve((size_t)R * D * 2);
    bf16_t* hb   = (bf16_t*)carve((size_t)R * M4 * 2);
    float*  sc   = (float*) carve((size_t)B * H * S * S * 4);   // 96 MB
    bf16_t* awb  = (bf16_t*)carve((size_t)B * H * S * S * 2);   // 48 MB

    auto cvt = [&](const float* src, bf16_t* dst, long n) {
        k_cvt_bf16<<<dim3((unsigned)((n + 255) / 256)), dim3(256), 0, stream>>>(src, dst, n);
    };

    // ---- weights to bf16 ----
    cvt(wq, wqb, (long)D * D);
    cvt(wk, wkb, (long)D * D);
    cvt(wv, wvb, (long)D * D);
    cvt(wo, wob, (long)D * D);
    cvt(w1, w1b, (long)D * M4);
    cvt(w2, w2b, (long)M4 * D);

    // ---- LN1 ----
    k_layernorm<<<dim3((unsigned)R), dim3(256), 0, stream>>>(x, ln1g, ln1b, ylnb);

    // ---- QKV projections (bf16 out); q scale folded into scores GEMM ----
    dim3 blk(256);
    dim3 gProj(D / 64, (unsigned)(R / 128), 1);
    k_gemm<false, false, false, true><<<gProj, blk, 0, stream>>>(
        ylnb, D, 0, 0, wqb, D, 0, 0, bq, nullptr, nullptr, qb, D, 0, 0, 1, D, 1.0f);
    k_gemm<false, false, false, true><<<gProj, blk, 0, stream>>>(
        ylnb, D, 0, 0, wkb, D, 0, 0, bk, nullptr, nullptr, kb, D, 0, 0, 1, D, 1.0f);
    k_gemm<false, false, false, true><<<gProj, blk, 0, stream>>>(
        ylnb, D, 0, 0, wvb, D, 0, 0, bv, nullptr, nullptr, vb, D, 0, 0, 1, D, 1.0f);

    // ---- scores[b,h] = (q/8) @ k^T : M=S, N=S, K=Dh, batched over B*H ----
    {
        dim3 g(S / 64, S / 128, (unsigned)(B * H));
        k_gemm<true, false, false, false><<<g, blk, 0, stream>>>(
            qb, D, (long)S * D, 64,
            kb, D, (long)S * D, 64,
            nullptr, nullptr,
            sc, nullptr, S, (long)H * S * S, (long)S * S,
            H, 64, 0.125f);
    }

    // ---- poly_norm per score row ----
    k_poly_norm<<<dim3((unsigned)((long)B * H * S)), dim3(256), 0, stream>>>(sc, awb);

    // ---- attn[b,h] = attn_w @ v : M=S, N=Dh, K=S ----
    {
        dim3 g(1, S / 128, (unsigned)(B * H));
        k_gemm<false, false, false, true><<<g, blk, 0, stream>>>(
            awb, S, (long)H * S * S, (long)S * S,
            vb, D, (long)S * D, 64,
            nullptr, nullptr,
            nullptr, attb, D, (long)S * D, 64,
            H, S, 1.0f);
    }

    // ---- x1 = x + attn @ wo + bo ----
    k_gemm<false, false, true, false><<<gProj, blk, 0, stream>>>(
        attb, D, 0, 0, wob, D, 0, 0, bo, x, x1, nullptr, D, 0, 0, 1, D, 1.0f);

    // ---- LN2 ----
    k_layernorm<<<dim3((unsigned)R), dim3(256), 0, stream>>>(x1, ln2g, ln2b, y2b);

    // ---- h = gelu(y2 @ w1 + b1) ----
    {
        dim3 g(M4 / 64, (unsigned)(R / 128), 1);
        k_gemm<false, true, false, true><<<g, blk, 0, stream>>>(
            y2b, D, 0, 0, w1b, M4, 0, 0, b1, nullptr, nullptr, hb, M4, 0, 0, 1, D, 1.0f);
    }

    // ---- out = x1 + h @ w2 + b2 ----
    k_gemm<false, false, true, false><<<gProj, blk, 0, stream>>>(
        hb, M4, 0, 0, w2b, D, 0, 0, b2, x1, (float*)d_out, nullptr, D, 0, 0, 1, M4, 1.0f);
}
